// FNN_v2_1460288880844
// MI455X (gfx1250) — matrix operand order repaired and checked by reading, not compiled, not run on hardware
//
#include <hip/hip_runtime.h>
#include <hip/hip_bf16.h>

typedef __attribute__((ext_vector_type(16))) _Float16 v16h;
typedef __attribute__((ext_vector_type(8)))  _Float16 v8h;
typedef __attribute__((ext_vector_type(4)))  _Float16 v4h;
typedef __attribute__((ext_vector_type(8)))  float    v8f;

#define NEG_SLOPE 0.01f
#define LN_EPS    1e-5f

static __device__ __forceinline__ v16h cat8(v8h lo, v8h hi) {
    return __builtin_shufflevector(lo, hi, 0,1,2,3,4,5,6,7,8,9,10,11,12,13,14,15);
}
static __device__ __forceinline__ float lrelu(float v) {
    return fmaxf(v, NEG_SLOPE * v);   // exact leaky relu, single v_max_num_f32
}

// ---------------------------------------------------------------------------
// Kernel 0: zero the segment-sum buckets (graph-replay safe re-init)
// ---------------------------------------------------------------------------
__global__ void zero_f32(float* p, int n) {
    int i = blockIdx.x * blockDim.x + threadIdx.x;
    if (i < n) p[i] = 0.0f;
}

// ---------------------------------------------------------------------------
// Kernel 1: per-edge MLP 3->64->64->64->1, all three matmuls on WMMA.
//  - layer 0: input augmented to [xa,xb,xc,1] (bias folded), K=32 zero-padded
//  - LayerNorm folded into the NEXT layer's weights (exact):
//       norm(v)@W^T = inv*(v @ (W*gamma)^T) + (-mu*inv)*u + t
//  - Feature permutation pi(t*16+n)=4n+t: lane's 4 outputs per D-row are
//    LDS-contiguous -> one ds_store_b64 (packed f16x4); the next layer's
//    weight columns are permuted identically at init (matmul invariant).
//  - Software pipelining: layer N+1's WMMAs depend only on the stored raw
//    activations, NOT on layer N's LN stats, so the shuffle-reduction trees
//    run in the shadow of the next layer's WMMAs.
// One wave32 owns a 16-edge tile; 20 WMMAs per tile.
// ---------------------------------------------------------------------------
__global__ void __launch_bounds__(256)
edge_mlp_wmma(const int* __restrict__ src, const int* __restrict__ dst,
              const float* __restrict__ x_s, const float* __restrict__ x_t,
              const float* __restrict__ ea,
              const float* __restrict__ w0, const float* __restrict__ b0,
              const float* __restrict__ w1, const float* __restrict__ b1,
              const float* __restrict__ w2, const float* __restrict__ b2,
              const float* __restrict__ w3, const float* __restrict__ b3,
              const float* __restrict__ lnw, const float* __restrict__ lnb,
              float* __restrict__ yws, float* __restrict__ ssum, float* __restrict__ dsum,
              int E)
{
    __shared__ _Float16 W0a[64][40];      // [out][K=32 padded]: w0 | b0 | zeros
    __shared__ _Float16 W1p[64][72];      // w1 * gamma0, columns permuted by pi
    __shared__ _Float16 W2p[64][72];      // w2 * gamma1, columns permuted by pi
    __shared__ _Float16 Ah0[8][16][40];   // per-wave layer-0 A tiles (K=32, zero-padded)
    __shared__ _Float16 Ah[8][16][72];    // per-wave activations [edge][pi(feat)]
    __shared__ float u1sh[64], t1sh[64], u2sh[64], t2sh[64], w3psh[64];
    __shared__ float u3t3[2];

    const int tid = threadIdx.x;
    // ---- one-time per-block weight prep -----------------------------------
    for (int i = tid; i < 64 * 64; i += 256) {
        const int r = i >> 6, c = i & 63;
        const int cp = ((c & 15) << 2) | (c >> 4);   // pi(c): contraction permute
        W1p[r][cp] = (_Float16)(w1[i] * lnw[c]);
        W2p[r][cp] = (_Float16)(w2[i] * lnw[64 + c]);
    }
    for (int i = tid; i < 64 * 32; i += 256) {
        const int r = i >> 5, k = i & 31;
        const float v = (k < 3) ? w0[r * 3 + k] : (k == 3 ? b0[r] : 0.f);
        W0a[r][k] = (_Float16)v;
    }
    for (int i = tid; i < 8 * 16 * 40; i += 256)
        ((_Float16*)Ah0)[i] = (_Float16)0.f;        // K=4..31 stay zero forever
    if (tid < 64) {
        float ua = 0.f, ta = 0.f, ub = 0.f, tb = 0.f;
        #pragma unroll 4
        for (int f = 0; f < 64; ++f) {
            const float a = w1[tid * 64 + f], b = w2[tid * 64 + f];
            ua += a * lnw[f];       ta += a * lnb[f];
            ub += b * lnw[64 + f];  tb += b * lnb[64 + f];
        }
        u1sh[tid] = ua; t1sh[tid] = ta + b1[tid];
        u2sh[tid] = ub; t2sh[tid] = tb + b2[tid];
        w3psh[tid] = w3[tid] * lnw[128 + tid];
    }
    if (tid == 0) {
        float u = 0.f, t = 0.f;
        #pragma unroll 4
        for (int f = 0; f < 64; ++f) {
            u += w3[f] * lnw[128 + f];
            t += w3[f] * lnb[128 + f];
        }
        u3t3[0] = u; u3t3[1] = t + b3[0];
    }
    __syncthreads();

    const int lane = tid & 31;
    const int wave = tid >> 5;
    const int hsel = lane >> 4;     // 0 = lanes 0-15, 1 = lanes 16-31
    const int n    = lane & 15;
    const float u3 = u3t3[0], t3 = u3t3[1];

    const int tilesTotal  = (E + 15) >> 4;
    const int wavesGlobal = gridDim.x * 8;

    for (int tile = blockIdx.x * 8 + wave; tile < tilesTotal; tile += wavesGlobal) {
        const int e0 = tile << 4;
        const int eMine = e0 + n;
        const bool ok = eMine < E;
        const int ec = ok ? eMine : (E - 1);
        const int si = src[ec], di = dst[ec];
        const float xa = x_s[2 * si + 1];
        const float xb = x_t[2 * di + 1];
        const float xc = ea[ec];

        // ---- layer 0: stage [xa,xb,xc,1] and matmul on WMMA --------------
        if (hsel == 0) {
            v4h in4 = { (_Float16)xa, (_Float16)xb, (_Float16)xc, (_Float16)1.0f };
            *(v4h*)(&Ah0[wave][n][0]) = in4;
        }
        const _Float16* arow0 = &Ah0[wave][n][0];
        v16h a00 = cat8(*(const v8h*)(arow0 + hsel * 16),
                        *(const v8h*)(arow0 + 8 + hsel * 16));
        v8f acc0[4];
        #pragma unroll
        for (int t = 0; t < 4; ++t) {
            const _Float16* wr = &W0a[t * 16 + n][hsel * 16];
            v16h bf = cat8(*(const v8h*)wr, *(const v8h*)(wr + 8));
            v8f z = {};
            acc0[t] = __builtin_amdgcn_wmma_f32_16x16x32_f16(
                          false, a00, false, bf, (short)0, z, false, false);
        }

        // epilogue 0: LeakyReLU, packed raw store, fill stats arrays
        float s[8], q[8], iv[8], st[8];
        #pragma unroll
        for (int r = 0; r < 8; ++r) {
            const float v0 = lrelu(acc0[0][r]);
            const float v1 = lrelu(acc0[1][r]);
            const float v2 = lrelu(acc0[2][r]);
            const float v3 = lrelu(acc0[3][r]);
            v4h pk = { (_Float16)v0, (_Float16)v1, (_Float16)v2, (_Float16)v3 };
            *(v4h*)(&Ah[wave][r + 8 * hsel][n * 4]) = pk;
            s[r] = (v0 + v1) + (v2 + v3);
            q[r] = (v0 * v0 + v1 * v1) + (v2 * v2 + v3 * v3);
        }

        float dot[8];
        float w3l[4];
        #pragma unroll
        for (int t = 0; t < 4; ++t) w3l[t] = w3psh[t * 16 + n];

        // ---- layers 1 & 2: WMMAs issued BEFORE the previous layer's stats
        // reduction, so the ds_bpermute trees run in the WMMA shadow.
        #pragma unroll
        for (int L = 0; L < 2; ++L) {
            const _Float16* arow = &Ah[wave][n][0];
            v16h A0 = cat8(*(const v8h*)(arow +  0 + hsel * 16),
                           *(const v8h*)(arow +  8 + hsel * 16));
            v16h A1 = cat8(*(const v8h*)(arow + 32 + hsel * 16),
                           *(const v8h*)(arow + 40 + hsel * 16));
            v8f acc[4];
            #pragma unroll
            for (int t = 0; t < 4; ++t) {
                const _Float16* wr0 = (L == 0) ? &W1p[t * 16 + n][hsel * 16]
                                               : &W2p[t * 16 + n][hsel * 16];
                const _Float16* wr1 = wr0 + 32;
                v16h B0 = cat8(*(const v8h*)wr0, *(const v8h*)(wr0 + 8));
                v16h B1 = cat8(*(const v8h*)wr1, *(const v8h*)(wr1 + 8));
                v8f z = {};
                v8f c = __builtin_amdgcn_wmma_f32_16x16x32_f16(
                            false, A0, false, B0, (short)0, z, false, false);
                acc[t] = __builtin_amdgcn_wmma_f32_16x16x32_f16(
                            false, A1, false, B1, (short)0, c, false, false);
            }

            // previous layer's LN stats reduction (overlaps WMMA execution);
            // independent temps -> scheduler can batch permutes per wait
            #pragma unroll
            for (int r = 0; r < 8; ++r) {
                float sr = s[r], qr = q[r];
                #pragma unroll
                for (int m = 1; m < 16; m <<= 1) {
                    const float ts = __shfl_xor(sr, m, 32);
                    const float tq = __shfl_xor(qr, m, 32);
                    sr += ts; qr += tq;
                }
                const float mu = sr * (1.f / 64.f);
                const float in = rsqrtf(qr * (1.f / 64.f) - mu * mu + LN_EPS);
                iv[r] = in; st[r] = -mu * in;
            }

            float ul[4], tl[4];
            #pragma unroll
            for (int t = 0; t < 4; ++t) {
                const int f = t * 16 + n;
                ul[t] = (L == 0) ? u1sh[f] : u2sh[f];
                tl[t] = (L == 0) ? t1sh[f] : t2sh[f];
            }

            #pragma unroll
            for (int r = 0; r < 8; ++r) {
                // z = inv*(rawprev @ Wfold) + (-mu*inv)*u + t   (exact LN fold)
                const float v0 = lrelu(fmaf(acc[0][r], iv[r], fmaf(st[r], ul[0], tl[0])));
                const float v1 = lrelu(fmaf(acc[1][r], iv[r], fmaf(st[r], ul[1], tl[1])));
                const float v2 = lrelu(fmaf(acc[2][r], iv[r], fmaf(st[r], ul[2], tl[2])));
                const float v3 = lrelu(fmaf(acc[3][r], iv[r], fmaf(st[r], ul[3], tl[3])));
                if (L == 0) {
                    v4h pk = { (_Float16)v0, (_Float16)v1, (_Float16)v2, (_Float16)v3 };
                    *(v4h*)(&Ah[wave][r + 8 * hsel][n * 4]) = pk;
                } else {
                    dot[r] = fmaf(v0, w3l[0], fmaf(v1, w3l[1],
                             fmaf(v2, w3l[2], v3 * w3l[3])));
                }
                s[r] = (v0 + v1) + (v2 + v3);
                q[r] = (v0 * v0 + v1 * v1) + (v2 * v2 + v3 * v3);
            }
        }

        // ---- final: layer-2 stats + dot reduced together (3 chains/row) --
        const int myR = n - 8 * hsel;   // writers: lanes 0..7 and 24..31
        #pragma unroll
        for (int r = 0; r < 8; ++r) {
            float sr = s[r], qr = q[r], dr = dot[r];
            #pragma unroll
            for (int m = 1; m < 16; m <<= 1) {
                const float ts = __shfl_xor(sr, m, 32);
                const float tq = __shfl_xor(qr, m, 32);
                const float td = __shfl_xor(dr, m, 32);
                sr += ts; qr += tq; dr += td;
            }
            if (myR == r) {
                const float mu = sr * (1.f / 64.f);
                const float in = rsqrtf(qr * (1.f / 64.f) - mu * mu + LN_EPS);
                float yv = fmaf(dr, in, fmaf(-mu * in, u3, t3));
                yv = fmaxf(yv, 0.f);
                if (ok) {
                    yws[eMine] = yv;
                    atomicAdd(&ssum[si], yv);
                    atomicAdd(&dsum[di], yv);
                }
            }
        }
    }
}

// ---------------------------------------------------------------------------
// Kernel 2: per-edge coef MLP 4 -> 32 (ReLU, LayerNorm) -> 1 (ReLU); out = y*coef
// ---------------------------------------------------------------------------
__global__ void __launch_bounds__(256)
coef_kernel(const int* __restrict__ src, const int* __restrict__ dst,
            const float* __restrict__ x_s, const float* __restrict__ x_t,
            const float* __restrict__ fw1, const float* __restrict__ fb1,
            const float* __restrict__ flnw, const float* __restrict__ flnb,
            const float* __restrict__ fw2, const float* __restrict__ fb2,
            const float* __restrict__ yws, const float* __restrict__ ssum,
            const float* __restrict__ dsum, float* __restrict__ out, int E)
{
    __shared__ float W[128], B[32], LW[32], LB[32], W2[32];
    __shared__ float b2s;
    const int tid = threadIdx.x;
    if (tid < 128) W[tid] = fw1[tid];
    if (tid < 32) { B[tid] = fb1[tid]; LW[tid] = flnw[tid]; LB[tid] = flnb[tid]; W2[tid] = fw2[tid]; }
    if (tid == 0) b2s = fb2[0];
    __syncthreads();

    const int e = blockIdx.x * blockDim.x + tid;
    if (e >= E) return;
    const int si = src[e], di = dst[e];
    const float c0 = x_s[2 * si + 1];
    const float c1 = ssum[si];
    const float c2 = x_t[2 * di + 1];
    const float c3 = dsum[di];

    float h[32];
    float s = 0.f, q = 0.f;
    #pragma unroll
    for (int j = 0; j < 32; ++j) {
        float v = W[j * 4 + 0] * c0 + W[j * 4 + 1] * c1 +
                  W[j * 4 + 2] * c2 + W[j * 4 + 3] * c3 + B[j];
        v = fmaxf(v, 0.f);
        h[j] = v; s += v; q += v * v;
    }
    const float mu = s * (1.f / 32.f);
    const float ivn = rsqrtf(q * (1.f / 32.f) - mu * mu + LN_EPS);
    float acc = b2s;
    #pragma unroll
    for (int j = 0; j < 32; ++j)
        acc += ((h[j] - mu) * ivn * LW[j] + LB[j]) * W2[j];
    acc = fmaxf(acc, 0.f);
    out[e] = yws[e] * acc;
}

// ---------------------------------------------------------------------------
extern "C" void kernel_launch(void* const* d_in, const int* in_sizes, int n_in,
                              void* d_out, int out_size, void* d_ws, size_t ws_size,
                              hipStream_t stream)
{
    const float* x_s  = (const float*)d_in[0];
    const float* x_t  = (const float*)d_in[1];
    const int*   ei   = (const int*)  d_in[2];
    const float* ea   = (const float*)d_in[3];
    const float* w0   = (const float*)d_in[4];
    const float* b0   = (const float*)d_in[5];
    const float* w1   = (const float*)d_in[6];
    const float* b1   = (const float*)d_in[7];
    const float* w2   = (const float*)d_in[8];
    const float* b2   = (const float*)d_in[9];
    const float* w3   = (const float*)d_in[10];
    const float* b3   = (const float*)d_in[11];
    const float* lnw  = (const float*)d_in[12];
    const float* lnb  = (const float*)d_in[13];
    const float* fw1  = (const float*)d_in[14];
    const float* fb1  = (const float*)d_in[15];
    const float* flnw = (const float*)d_in[16];
    const float* flnb = (const float*)d_in[17];
    const float* fw2  = (const float*)d_in[18];
    const float* fb2  = (const float*)d_in[19];

    const int E  = in_sizes[3];
    const int NS = in_sizes[0] / 2;
    const int NT = in_sizes[1] / 2;
    const int* src = ei;
    const int* dst = ei + E;

    float* yws  = (float*)d_ws;     // [E]
    float* ssum = yws + E;          // [NS]
    float* dsum = ssum + NS;        // [NT]
    float* out  = (float*)d_out;

    const int nz = NS + NT;
    zero_f32<<<(nz + 255) / 256, 256, 0, stream>>>(ssum, nz);

    const int tiles = (E + 15) / 16;
    int blocks = (tiles + 7) / 8;
    if (blocks > 1024) blocks = 1024;
    edge_mlp_wmma<<<blocks, 256, 0, stream>>>(src, dst, x_s, x_t, ea,
                                              w0, b0, w1, b1, w2, b2, w3, b3,
                                              lnw, lnb, yws, ssum, dsum, E);

    coef_kernel<<<(E + 255) / 256, 256, 0, stream>>>(src, dst, x_s, x_t,
                                                     fw1, fb1, flnw, flnb, fw2, fb2,
                                                     yws, ssum, dsum, out, E);
}
